// toeken_proj_56332791054480
// MI455X (gfx1250) — matrix operand order repaired and checked by reading, not compiled, not run on hardware
//
#include <hip/hip_runtime.h>
#include <hip/hip_bf16.h>

// Problem constants
#define Bc 8
#define Nn 4096
#define Mm 1024
#define Dd 128
#define Hh 512

typedef __attribute__((ext_vector_type(16))) __bf16 bf16x16;
typedef __attribute__((ext_vector_type(8)))  float  f32x8;

// Native bf16 conversion (backend emits v_cvt_pk_bf16_f32; RNE)
__device__ __forceinline__ unsigned short f32_to_bf16(float x) {
    union { __bf16 h; unsigned short s; } u;
    u.h = (__bf16)x;
    return u.s;
}

// Convert 8 f32 -> 8 bf16 packed in one uint4
__device__ __forceinline__ uint4 cvt8_bf16(const float4 a, const float4 b) {
    union { unsigned short s[8]; uint4 u; } r;
    r.s[0] = f32_to_bf16(a.x); r.s[1] = f32_to_bf16(a.y);
    r.s[2] = f32_to_bf16(a.z); r.s[3] = f32_to_bf16(a.w);
    r.s[4] = f32_to_bf16(b.x); r.s[5] = f32_to_bf16(b.y);
    r.s[6] = f32_to_bf16(b.z); r.s[7] = f32_to_bf16(b.w);
    return r.u;
}

// ---- CDNA5 async global->LDS copy (CDNA5 ISA), GVS mode:
// mem = SGPR base + 32-bit VGPR byte offset. Tracked by ASYNCcnt.
__device__ __forceinline__ void async_copy_b128(void* lds, const void* base, unsigned byte_off) {
    unsigned loff = (unsigned)(unsigned long long)lds;  // LDS offset = low 32 bits of generic addr
    asm volatile("global_load_async_to_lds_b128 %0, %1, %2"
                 :: "v"(loff), "v"(byte_off), "s"(base) : "memory");
}
// Wait until at most `imm` async ops remain outstanding (in-order completion).
#define WAIT_ASYNCCNT(imm) asm volatile("s_wait_asynccnt " imm ::: "memory")

// A fragment (16x32 bf16, M x K) from a row-major LDS tile.
// ISA 7.12.2: lanes 0-15 row M=lane, K 0-7 then 16-23; lanes 16-31 K 8-15 then 24-31.
__device__ __forceinline__ bf16x16 load_frag_a(const unsigned short* buf, int stride, int lane) {
    int r = lane & 15;
    int h = (lane < 16) ? 0 : 8;
    const unsigned short* p = buf + r * stride + h;
    union { uint4 u[2]; bf16x16 v; } f;
    f.u[0] = *(const uint4*)(p);
    f.u[1] = *(const uint4*)(p + 16);
    return f.v;
}

// B fragment (32x16 bf16, K x N) from an N-major LDS tile (row = n, col = k).
__device__ __forceinline__ bf16x16 load_frag_b(const unsigned short* buf, int stride, int lane) {
    int n = lane & 15;
    int kb = (lane < 16) ? 0 : 8;
    const unsigned short* p = buf + n * stride + kb;
    union { uint4 u[2]; bf16x16 v; } f;
    f.u[0] = *(const uint4*)(p);
    f.u[1] = *(const uint4*)(p + 16);
    return f.v;
}

__device__ __forceinline__ f32x8 wmma_bf16(bf16x16 a, bf16x16 b, f32x8 c) {
    return __builtin_amdgcn_wmma_f32_16x16x32_bf16(false, a, false, b, (short)0, c, false, false);
}

// ---------------------------------------------------------------------------
// Kernel 1: out_bf16[R x C] = bf16( A_f32[R x 128] @ W_f32[128 x C] + bias[C] )
// WG = 256 thr (8 waves). Tile 128(rows) x 64(cols); wave = 32x32 (2x2 WMMA).
// ---------------------------------------------------------------------------
__global__ __launch_bounds__(256) void proj_gemm_kernel(
    const float* __restrict__ A, const float* __restrict__ W,
    const float* __restrict__ bias, unsigned short* __restrict__ out, int C) {
    __shared__ __align__(16) unsigned short smem[8192];  // 16 KB
    unsigned short* Abuf = smem;          // [128][32]
    unsigned short* Wt   = smem + 4096;   // [64][32] N-major
    unsigned short* Cbuf = smem;          // [128][64] (epilogue reuse)

    const int tid  = threadIdx.x;
    const int lane = tid & 31;
    const int wave = tid >> 5;
    const int wm   = wave & 3;   // 32-row group
    const int wn   = wave >> 2;  // 32-col group
    const int row0 = blockIdx.y * 128;
    const int col0 = blockIdx.x * 64;

    f32x8 acc[2][2];
    #pragma unroll
    for (int s = 0; s < 2; ++s)
        #pragma unroll
        for (int t = 0; t < 2; ++t) acc[s][t] = 0.0f;

    for (int kk = 0; kk < 128; kk += 32) {
        // Stage A tile 128x32: b128 global loads, cvt, one b128 LDS store
        #pragma unroll
        for (int i = 0; i < 2; ++i) {
            int e0 = (i * 256 + tid) * 8;
            int r = e0 >> 5, c = e0 & 31;
            const float* gp = A + (unsigned)((row0 + r) * 128 + kk + c);
            float4 f0 = *(const float4*)(gp);
            float4 f1 = *(const float4*)(gp + 4);
            *(uint4*)&Abuf[r * 32 + c] = cvt8_bf16(f0, f1);
        }
        // Stage W tile 32x64 transposed into N-major
        #pragma unroll
        for (int i = 0; i < 8; ++i) {
            int e = i * 256 + tid;
            int kw = e >> 6, c = e & 63;
            Wt[c * 32 + kw] = f32_to_bf16(W[(unsigned)((kk + kw) * C + col0 + c)]);
        }
        __syncthreads();

        bf16x16 a0 = load_frag_a(&Abuf[(wm * 32 + 0) * 32], 32, lane);
        bf16x16 a1 = load_frag_a(&Abuf[(wm * 32 + 16) * 32], 32, lane);
        bf16x16 b0 = load_frag_b(&Wt[(wn * 32 + 0) * 32], 32, lane);
        bf16x16 b1 = load_frag_b(&Wt[(wn * 32 + 16) * 32], 32, lane);
        acc[0][0] = wmma_bf16(a0, b0, acc[0][0]);
        acc[0][1] = wmma_bf16(a0, b1, acc[0][1]);
        acc[1][0] = wmma_bf16(a1, b0, acc[1][0]);
        acc[1][1] = wmma_bf16(a1, b1, acc[1][1]);
        __syncthreads();
    }

    // Epilogue: +bias, cvt, transpose via LDS, coalesced b128 stores
    #pragma unroll
    for (int s = 0; s < 2; ++s)
        #pragma unroll
        for (int t = 0; t < 2; ++t) {
            int lc = wn * 32 + t * 16 + (lane & 15);
            float bv = bias[col0 + lc];
            int lr0 = wm * 32 + s * 16 + ((lane < 16) ? 0 : 8);
            #pragma unroll
            for (int v = 0; v < 8; ++v)
                Cbuf[(lr0 + v) * 64 + lc] = f32_to_bf16(acc[s][t][v] + bv);
        }
    __syncthreads();
    #pragma unroll
    for (int i = 0; i < 4; ++i) {
        int u = i * 256 + tid;           // 1024 uint4
        int r = u >> 3, c8 = (u & 7) * 8;
        *(uint4*)&out[(unsigned)((row0 + r) * C + col0 + c8)] = ((const uint4*)Cbuf)[u];
    }
}

// ---------------------------------------------------------------------------
// Kernel 2: colsum[b,m] = sum_n exp( q[b,n,:] . k[b,m,:] )
// S-GEMM with double-buffered async global->LDS staging (hh chunks of 32).
// ---------------------------------------------------------------------------
__global__ __launch_bounds__(256) void colsum_kernel(
    const unsigned short* __restrict__ q, const unsigned short* __restrict__ k,
    float* __restrict__ colsum) {
    __shared__ __align__(16) unsigned short smem[16384];   // 32 KB
    // ping-pong: Q buffer p at smem + p*4096, K buffer p at smem + 8192 + p*4096

    const int tid = threadIdx.x, lane = tid & 31, wave = tid >> 5;
    const int wn = wave & 3;   // 32-row (n) group
    const int wm = wave >> 2;  // 64-col (m) group
    const int b    = blockIdx.y;
    const int row0 = blockIdx.x * 128;
    const unsigned short* qb = q + (size_t)b * Nn * Hh;
    const unsigned short* kb = k + (size_t)b * Mm * Hh;

    for (int mt = 0; mt < Mm; mt += 128) {
        if (mt + 128 < Mm)   // hint: next m-tile's k rows -> global_prefetch_b8
            __builtin_prefetch(kb + (unsigned)((mt + 128 + (tid >> 1)) * Hh + (tid & 1) * 256), 0, 1);

        f32x8 s_acc[2][4];
        #pragma unroll
        for (int s = 0; s < 2; ++s)
            #pragma unroll
            for (int t = 0; t < 4; ++t) s_acc[s][t] = 0.0f;

        // issue stage: 4 async b128 per thread into ping-pong buffer p
        auto issue_stage = [&](int p, int hh) {
            unsigned short* Qp = smem + p * 4096;
            unsigned short* Kp = smem + 8192 + p * 4096;
            #pragma unroll
            for (int i = 0; i < 2; ++i) {
                int u = i * 256 + tid;
                int r = u >> 2, c = u & 3;
                async_copy_b128(Qp + u * 8, qb, (unsigned)((row0 + r) * Hh + hh + c * 8) * 2u);
            }
            #pragma unroll
            for (int i = 0; i < 2; ++i) {
                int u = i * 256 + tid;
                int r = u >> 2, c = u & 3;
                async_copy_b128(Kp + u * 8, kb, (unsigned)((mt + r) * Hh + hh + c * 8) * 2u);
            }
        };

        issue_stage(0, 0);
        for (int st = 0; st < 16; ++st) {      // 16 stages of K=32
            if (st + 1 < 16) {
                issue_stage((st + 1) & 1, (st + 1) * 32);
                WAIT_ASYNCCNT("0x4");          // older stage done; newer 4 still in flight
            } else {
                WAIT_ASYNCCNT("0x0");
            }
            __syncthreads();
            const unsigned short* Qp = smem + (st & 1) * 4096;
            const unsigned short* Kp = smem + 8192 + (st & 1) * 4096;
            bf16x16 a0 = load_frag_a(&Qp[(wn * 32 + 0) * 32], 32, lane);
            bf16x16 a1 = load_frag_a(&Qp[(wn * 32 + 16) * 32], 32, lane);
            #pragma unroll
            for (int t = 0; t < 4; ++t) {
                bf16x16 bf = load_frag_b(&Kp[(wm * 64 + t * 16) * 32], 32, lane);
                s_acc[0][t] = wmma_bf16(a0, bf, s_acc[0][t]);
                s_acc[1][t] = wmma_bf16(a1, bf, s_acc[1][t]);
            }
            __syncthreads();
        }

        // exp + column partial sums: lanes L and L^16 share a column
        #pragma unroll
        for (int t = 0; t < 4; ++t) {
            float p = 0.0f;
            #pragma unroll
            for (int s = 0; s < 2; ++s)
                #pragma unroll
                for (int v = 0; v < 8; ++v) p += __expf(s_acc[s][t][v]);
            p += __shfl_xor(p, 16, 32);
            if (lane < 16)
                atomicAdd(&colsum[(unsigned)(b * Mm + mt + wm * 64 + t * 16 + lane)], p);
        }
    }
}

// ---------------------------------------------------------------------------
// Kernel 3: out[b,n,d] = Xc[b,n,d] + sum_m (exp(S)/colsum[m]) * v[b,m,d]
// ---------------------------------------------------------------------------
__global__ __launch_bounds__(256) void attn_out_kernel(
    const unsigned short* __restrict__ q, const unsigned short* __restrict__ k,
    const unsigned short* __restrict__ v, const float* __restrict__ colsum,
    const float* __restrict__ Xc, float* __restrict__ out) {
    __shared__ __align__(16) unsigned short smem[20480];    // 40 KB
    // ping-pong stage buffers: Q at smem + p*4096, K at smem + 8192 + p*4096
    unsigned short* Pbuf = smem;            // [128][128], overlaps all 4 stage buffers
    unsigned short* Vbuf = smem + 16384;    // [128 d][32 m], d-major

    const int tid = threadIdx.x, lane = tid & 31, wave = tid >> 5;
    const int wn = wave & 3;   // 32-row (n) group
    const int wm = wave >> 2;  // 64-col (m in phase 1 / d in phase 2) group
    const int b    = blockIdx.y;
    const int row0 = blockIdx.x * 128;
    const unsigned short* qb = q + (size_t)b * Nn * Hh;
    const unsigned short* kb = k + (size_t)b * Mm * Hh;
    const unsigned short* vb = v + (size_t)b * Mm * Dd;

    f32x8 o_acc[2][4];
    #pragma unroll
    for (int s = 0; s < 2; ++s)
        #pragma unroll
        for (int t = 0; t < 4; ++t) o_acc[s][t] = 0.0f;

    for (int mt = 0; mt < Mm; mt += 128) {
        if (mt + 128 < Mm)
            __builtin_prefetch(kb + (unsigned)((mt + 128 + (tid >> 1)) * Hh + (tid & 1) * 256), 0, 1);

        // ---- Phase 1: S tile 128(n) x 128(m) = q @ k^T, double-buffered ----
        f32x8 s_acc[2][4];
        #pragma unroll
        for (int s = 0; s < 2; ++s)
            #pragma unroll
            for (int t = 0; t < 4; ++t) s_acc[s][t] = 0.0f;

        auto issue_stage = [&](int p, int hh) {
            unsigned short* Qp = smem + p * 4096;
            unsigned short* Kp = smem + 8192 + p * 4096;
            #pragma unroll
            for (int i = 0; i < 2; ++i) {
                int u = i * 256 + tid;
                int r = u >> 2, c = u & 3;
                async_copy_b128(Qp + u * 8, qb, (unsigned)((row0 + r) * Hh + hh + c * 8) * 2u);
            }
            #pragma unroll
            for (int i = 0; i < 2; ++i) {
                int u = i * 256 + tid;
                int r = u >> 2, c = u & 3;
                async_copy_b128(Kp + u * 8, kb, (unsigned)((mt + r) * Hh + hh + c * 8) * 2u);
            }
        };

        issue_stage(0, 0);
        for (int st = 0; st < 16; ++st) {
            if (st + 1 < 16) {
                issue_stage((st + 1) & 1, (st + 1) * 32);
                WAIT_ASYNCCNT("0x4");
            } else {
                WAIT_ASYNCCNT("0x0");
            }
            __syncthreads();
            const unsigned short* Qp = smem + (st & 1) * 4096;
            const unsigned short* Kp = smem + 8192 + (st & 1) * 4096;
            bf16x16 a0 = load_frag_a(&Qp[(wn * 32 + 0) * 32], 32, lane);
            bf16x16 a1 = load_frag_a(&Qp[(wn * 32 + 16) * 32], 32, lane);
            #pragma unroll
            for (int t = 0; t < 4; ++t) {
                bf16x16 bf = load_frag_b(&Kp[(wm * 64 + t * 16) * 32], 32, lane);
                s_acc[0][t] = wmma_bf16(a0, bf, s_acc[0][t]);
                s_acc[1][t] = wmma_bf16(a1, bf, s_acc[1][t]);
            }
            __syncthreads();
        }

        // ---- P = exp(S) / colsum, bf16 into LDS (row-major n x m) ----
        #pragma unroll
        for (int t = 0; t < 4; ++t) {
            int cm = wm * 64 + t * 16 + (lane & 15);
            float rcp = 1.0f / colsum[(unsigned)(b * Mm + mt + cm)];
            #pragma unroll
            for (int s = 0; s < 2; ++s) {
                int rb = wn * 32 + s * 16 + ((lane < 16) ? 0 : 8);
                #pragma unroll
                for (int vv = 0; vv < 8; ++vv)
                    Pbuf[(rb + vv) * 128 + cm] = f32_to_bf16(__expf(s_acc[s][t][vv]) * rcp);
            }
        }
        __syncthreads();

        // ---- Phase 2: o_acc += P[128n x 128m] @ v[128m x 128d] ----
        for (int mk = 0; mk < 128; mk += 32) {
            #pragma unroll
            for (int i = 0; i < 2; ++i) {     // stage v chunk transposed: [128 d][32 m]
                int u = i * 256 + tid;
                int mm = u >> 4, ch = u & 15;
                union { uint4 u4; unsigned short s[8]; } val;
                val.u4 = *(const uint4*)(vb + (unsigned)((mt + mk + mm) * Dd + ch * 8));
                #pragma unroll
                for (int j = 0; j < 8; ++j)
                    Vbuf[(ch * 8 + j) * 32 + mm] = val.s[j];
            }
            __syncthreads();
            bf16x16 a0 = load_frag_a(&Pbuf[(wn * 32 + 0) * 128 + mk], 128, lane);
            bf16x16 a1 = load_frag_a(&Pbuf[(wn * 32 + 16) * 128 + mk], 128, lane);
            #pragma unroll
            for (int t = 0; t < 4; ++t) {
                bf16x16 bf = load_frag_b(&Vbuf[(wm * 64 + t * 16) * 32], 32, lane);
                o_acc[0][t] = wmma_bf16(a0, bf, o_acc[0][t]);
                o_acc[1][t] = wmma_bf16(a1, bf, o_acc[1][t]);
            }
            __syncthreads();
        }
    }

    // ---- Epilogue: out = Xc + o_acc, transposed via LDS, float4 stores ----
    // Each wave's d-range (wm*64..wm*64+63) is exactly one 64-col half.
    const float* xb = Xc + ((size_t)b * Nn + row0) * Dd;
    float* ob = out + ((size_t)b * Nn + row0) * Dd;
    float* Cf = (float*)smem;   // [128][64] f32 = 32 KB (reuses P region)
    #pragma unroll
    for (int half = 0; half < 2; ++half) {
        if (wm == half) {
            #pragma unroll
            for (int t = 0; t < 4; ++t) {
                int lc = t * 16 + (lane & 15);
                #pragma unroll
                for (int s = 0; s < 2; ++s) {
                    int lr0 = wn * 32 + s * 16 + ((lane < 16) ? 0 : 8);
                    #pragma unroll
                    for (int vv = 0; vv < 8; ++vv)
                        Cf[(lr0 + vv) * 64 + lc] = o_acc[s][t][vv];
                }
            }
        }
        __syncthreads();
        #pragma unroll
        for (int i = 0; i < 8; ++i) {
            int u = i * 256 + tid;          // 2048 float4 (128 rows x 16 float4)
            int r = u >> 4, c4 = (u & 15) * 4;
            unsigned goff = (unsigned)(r * Dd + half * 64 + c4);
            float4 x = *(const float4*)(xb + goff);
            float4 cv = ((const float4*)Cf)[u];
            float4 o;
            o.x = x.x + cv.x; o.y = x.y + cv.y; o.z = x.z + cv.z; o.w = x.w + cv.w;
            *(float4*)(ob + goff) = o;
        }
        __syncthreads();
    }
}

// ---------------------------------------------------------------------------
extern "C" void kernel_launch(void* const* d_in, const int* in_sizes, int n_in,
                              void* d_out, int out_size, void* d_ws, size_t ws_size,
                              hipStream_t stream) {
    (void)in_sizes; (void)n_in; (void)out_size;
    const float* Xc   = (const float*)d_in[0];
    const float* Tout = (const float*)d_in[1];
    const float* W1   = (const float*)d_in[2];
    const float* b1   = (const float*)d_in[3];
    const float* W2   = (const float*)d_in[4];
    const float* b2   = (const float*)d_in[5];
    const float* W3   = (const float*)d_in[6];
    const float* b3   = (const float*)d_in[7];

    // Workspace layout (bf16 intermediates, all L2-resident: ~42 MB total)
    unsigned short* qws = (unsigned short*)d_ws;                 // [B*N, H]
    unsigned short* kws = qws + (size_t)Bc * Nn * Hh;            // [B*M, H]
    unsigned short* vws = kws + (size_t)Bc * Mm * Hh;            // [B*M, D]
    float* colsum = (float*)(vws + (size_t)Bc * Mm * Dd);        // [B, M]
    (void)ws_size;

    (void)hipMemsetAsync(colsum, 0, (size_t)Bc * Mm * sizeof(float), stream);

    // q = Xc @ W1 + b1   (32768 x 128) x (128 x 512)
    proj_gemm_kernel<<<dim3(Hh / 64, (Bc * Nn) / 128), 256, 0, stream>>>(Xc, W1, b1, qws, Hh);
    // k = Tout @ W2 + b2 (8192 x 128) x (128 x 512)
    proj_gemm_kernel<<<dim3(Hh / 64, (Bc * Mm) / 128), 256, 0, stream>>>(Tout, W2, b2, kws, Hh);
    // v = Tout @ W3 + b3 (8192 x 128) x (128 x 128)
    proj_gemm_kernel<<<dim3(Dd / 64, (Bc * Mm) / 128), 256, 0, stream>>>(Tout, W3, b3, vws, Dd);

    // colsum[b,m] = sum_n exp(S)
    colsum_kernel<<<dim3(Nn / 128, Bc), 256, 0, stream>>>(qws, kws, colsum);

    // out = Xc + (exp(S)/colsum) @ v
    attn_out_kernel<<<dim3(Nn / 128, Bc), 256, 0, stream>>>(qws, kws, vws, colsum, Xc, (float*)d_out);
}
